// TalkingHeadAttn_57793079935700
// MI455X (gfx1250) — matrix operand order repaired and checked by reading, not compiled, not run on hardware
//
#include <hip/hip_runtime.h>
#include <hip/hip_bf16.h>

// ---------------------------------------------------------------------------
// Talking-heads attention forward for MI455X (gfx1250, wave32, WMMA).
// bf16 operands / f32 accumulation via v_wmma_f32_16x16x32_bf16.
// GEMMs blocked 16x64 per wave (A-fragment reused by 4 WMMAs).
// k_scores stages K-tiles into LDS via async global->LDS loads (ASYNCcnt).
// ---------------------------------------------------------------------------

typedef __attribute__((ext_vector_type(16))) __bf16 v16bf;
typedef __attribute__((ext_vector_type(8)))  __bf16 v8bf;
typedef __attribute__((ext_vector_type(8)))  float  v8f;
typedef int v4i_ __attribute__((vector_size(16)));

#define B_    64
#define N_    197
#define C_    512
#define H_    8
#define D_    64
#define M_    (B_ * N_)      // 12608 = 788 * 16 exactly
#define MT_   (M_ / 16)      // 788
#define QKVN  1536
#define NP16  208            // N padded to 16 (M/N tile dims)
#define NP32  224            // N padded to 32 (K dim of attn@V)

#define AS1 __attribute__((address_space(1)))
#define AS3 __attribute__((address_space(3)))

#if defined(__has_builtin)
#if __has_builtin(__builtin_amdgcn_global_load_async_to_lds_b128)
#define USE_ASYNC_LDS 1
#endif
#endif
#ifndef USE_ASYNC_LDS
#define USE_ASYNC_LDS 0
#endif

__device__ __forceinline__ void wait_asynccnt0() {
#if defined(__has_builtin)
#if __has_builtin(__builtin_amdgcn_s_wait_asynccnt)
  __builtin_amdgcn_s_wait_asynccnt(0);
  return;
#else
  asm volatile("s_wait_asynccnt 0x0" ::: "memory");
  return;
#endif
#else
  asm volatile("s_wait_asynccnt 0x0" ::: "memory");
#endif
}

// ---------------- WMMA fragment loaders (per CDNA5 ISA 7.12.2) --------------

// A-matrix 16x32 bf16, row-major source with leading dim `ld` (elements).
// lane L: row = L%16; K-chunks [8*(L/16) .. +7] and [16+8*(L/16) .. +7].
__device__ __forceinline__ v16bf load_a16x32(const __bf16* base, int ld, int lane) {
  const int row = lane & 15;
  const int kg  = (lane >> 4) << 3;          // 0 or 8
  const __bf16* p = base + (size_t)row * ld + kg;
  v8bf lo = *reinterpret_cast<const v8bf*>(p);
  v8bf hi = *reinterpret_cast<const v8bf*>(p + 16);
  v16bf r;
#pragma unroll
  for (int i = 0; i < 8; ++i) { r[i] = lo[i]; r[i + 8] = hi[i]; }
  return r;
}

// B-matrix 32x16 bf16 from a column-contiguous source: element B[k][n] lives
// at base + n*ld + k. lane L: col = L%16; K-chunks [8*(L/16) .. +7] and [16+8*(L/16) .. +7].
__device__ __forceinline__ v16bf load_b32x16(const __bf16* base, int ld, int lane) {
  const int col = lane & 15;
  const int kb  = (lane >> 4) << 3;          // 0 or 8
  const __bf16* p = base + (size_t)col * ld + kb;
  v8bf lo = *reinterpret_cast<const v8bf*>(p);
  v8bf hi = *reinterpret_cast<const v8bf*>(p + 16);
  v16bf r;
#pragma unroll
  for (int i = 0; i < 8; ++i) { r[i] = lo[i]; r[i + 8] = hi[i]; }
  return r;
}

__device__ __forceinline__ v8f wmma_bf16(v16bf a, v16bf b, v8f c) {
  return __builtin_amdgcn_wmma_f32_16x16x32_bf16(false, a, false, b,
                                                 (short)0, c, false, false);
}

// ---------------------------- utility kernels -------------------------------

__global__ void k_zero(uint4* p, size_t n16) {
  size_t i = (size_t)blockIdx.x * blockDim.x + threadIdx.x;
  const size_t stride = (size_t)gridDim.x * blockDim.x;
  const uint4 z = {0u, 0u, 0u, 0u};
  for (; i < n16; i += stride) p[i] = z;
}

__global__ void k_cvt_bf16(const float* __restrict__ x, __bf16* __restrict__ y, int n) {
  int i = blockIdx.x * blockDim.x + threadIdx.x;
  const int stride = gridDim.x * blockDim.x;
  for (; i < n; i += stride) y[i] = (__bf16)x[i];
}

// out[n*K + k] = in[k*Ncol + n]  (make columns contiguous for B-fragments)
__global__ void k_transpose_cvt(const float* __restrict__ in, __bf16* __restrict__ out,
                                int K, int Ncol) {
  int i = blockIdx.x * blockDim.x + threadIdx.x;
  const int stride = gridDim.x * blockDim.x;
  const int total = K * Ncol;
  for (; i < total; i += stride) {
    const int n = i / K, k = i % K;
    out[i] = (__bf16)in[(size_t)k * Ncol + n];
  }
}

// ------------------- stage 1: QKV GEMM + scatter to q/k/vT ------------------
// Y = xb(12608x512) @ w_qkv(512x1536) + b; one wave per 16x64 tile
// (A-fragment reused across 4 WMMAs).
__global__ __launch_bounds__(256) void k_qkv(const __bf16* __restrict__ xb,
                                             const __bf16* __restrict__ wT,
                                             const float* __restrict__ bqkv,
                                             __bf16* __restrict__ q,
                                             __bf16* __restrict__ kmat,
                                             __bf16* __restrict__ vt) {
  const int lane = threadIdx.x & 31;
  const int gw   = blockIdx.x * 8 + (threadIdx.x >> 5);
  const int tm   = gw / (QKVN / 64);   // 24 col-blocks of 64
  const int tn   = gw % (QKVN / 64);
  const __bf16* abase = xb + (size_t)tm * 16 * C_;
  const __bf16* bbase = wT + (size_t)tn * 64 * C_;
  v8f acc[4] = {};
#pragma unroll 2
  for (int s = 0; s < C_ / 32; ++s) {
    if (s + 1 < C_ / 32) {
      __builtin_prefetch(abase + (s + 1) * 32 + (size_t)(lane & 15) * C_, 0, 3);
      __builtin_prefetch(bbase + (s + 1) * 32 + (size_t)(lane & 15) * C_, 0, 3);
    }
    v16bf a = load_a16x32(abase + s * 32, C_, lane);
#pragma unroll
    for (int t = 0; t < 4; ++t) {
      v16bf b = load_b32x16(bbase + (size_t)t * 16 * C_ + s * 32, C_, lane);
      acc[t] = wmma_bf16(a, b, acc[t]);
    }
  }
  // 64-col block maps to a single (sec,h): 1536 = 3 * 8 * 64
  const int sec = tn >> 3;             // 0:q 1:k 2:v
  const int h   = tn & 7;
  const int r0  = (lane >> 4) << 3;
  int bi[8], tt[8];
#pragma unroll
  for (int r = 0; r < 8; ++r) {
    const int m = tm * 16 + r0 + r;    // token index, always < 12608
    bi[r] = m / N_;
    tt[r] = m % N_;
  }
#pragma unroll
  for (int t = 0; t < 4; ++t) {
    const int d    = t * 16 + (lane & 15);
    const float bias = bqkv[(sec * H_ + h) * D_ + d];
#pragma unroll
    for (int r = 0; r < 8; ++r) {
      const int bh = bi[r] * H_ + h;
      const float v = acc[t][r] + bias;
      if (sec == 0)      q   [((size_t)bh * NP16 + tt[r]) * D_ + d] = (__bf16)(v * 0.125f);
      else if (sec == 1) kmat[((size_t)bh * NP16 + tt[r]) * D_ + d] = (__bf16)v;
      else               vt  [((size_t)bh * D_  + d) * NP32 + tt[r]] = (__bf16)v;
    }
  }
}

// --------------------- stage 2: S = q @ k^T per (b,h) -----------------------
// Each wave stages its 16x64 K-tile (2 KB) into wave-private LDS via async
// global->LDS loads (ASYNCcnt), then reads B-fragments via ds loads.
__global__ __launch_bounds__(256) void k_scores(const __bf16* __restrict__ q,
                                                const __bf16* __restrict__ kmat,
                                                float* __restrict__ attnf) {
  __shared__ alignas(16) __bf16 kstage[8][16 * D_];   // 8 waves * 2 KB
  const int lane = threadIdx.x & 31;
  const int wav  = threadIdx.x >> 5;
  const int gw   = blockIdx.x * 8 + wav;
  const int bh   = gw / 169;           // 13*13 tiles
  const int t2   = gw % 169;
  const int ti   = t2 / 13, tj = t2 % 13;
  const __bf16* abase = q    + (size_t)bh * NP16 * D_ + (size_t)ti * 16 * D_;
  const __bf16* bbase = kmat + (size_t)bh * NP16 * D_ + (size_t)tj * 16 * D_;
  __bf16* kbuf = &kstage[wav][0];

  // stage the 16x64 bf16 k-tile: 128 chunks of 16B, 4 per lane
#if USE_ASYNC_LDS
#pragma unroll
  for (int it = 0; it < 4; ++it) {
    const int idx = it * 32 + lane;
    const int row = idx >> 3, c8 = idx & 7;
    __builtin_amdgcn_global_load_async_to_lds_b128(
        (AS1 v4i_*)(bbase + (size_t)row * D_ + c8 * 8),
        (AS3 v4i_*)(kbuf + idx * 8), 0, 0);
  }
  wait_asynccnt0();
#else
#pragma unroll
  for (int it = 0; it < 4; ++it) {
    const int idx = it * 32 + lane;
    const int row = idx >> 3, c8 = idx & 7;
    *reinterpret_cast<v8bf*>(kbuf + idx * 8) =
        *reinterpret_cast<const v8bf*>(bbase + (size_t)row * D_ + c8 * 8);
  }
#endif

  v8f acc = {};
#pragma unroll
  for (int s = 0; s < D_ / 32; ++s) {  // 2 steps
    v16bf a = load_a16x32(abase + s * 32, D_, lane);
    v16bf b = load_b32x16(kbuf + s * 32, D_, lane);  // ds_load_b128 from LDS
    acc = wmma_bf16(a, b, acc);
  }
  float* out = attnf + (size_t)bh * N_ * N_;
  const int j  = tj * 16 + (lane & 15);
  const int r0 = (lane >> 4) << 3;
  if (j < N_) {
#pragma unroll
    for (int r = 0; r < 8; ++r) {
      const int i = ti * 16 + r0 + r;
      if (i < N_) out[(size_t)i * N_ + j] = acc[r];
    }
  }
}

// ---- stage 3: talking-heads mix (w_l) -> softmax -> mix (w_w), per (b,i) ---
__global__ __launch_bounds__(256) void k_mix_softmax(float* __restrict__ attnf,
                                                     __bf16* __restrict__ attnb,
                                                     const float* __restrict__ wl,
                                                     const float* __restrict__ bl,
                                                     const float* __restrict__ ww,
                                                     const float* __restrict__ bw) {
  __shared__ float swl[64], sww[64], sbl[8], sbw[8];
  __shared__ float red[8][256];
  const int tid = threadIdx.x;
  if (tid < 64) { swl[tid] = wl[tid]; sww[tid] = ww[tid]; }
  if (tid < 8)  { sbl[tid] = bl[tid]; sbw[tid] = bw[tid]; }
  __syncthreads();

  const int b = blockIdx.x / N_;
  const int i = blockIdx.x % N_;
  const int j = tid;
  const bool act = (j < N_);

  float s[8];
#pragma unroll
  for (int h = 0; h < 8; ++h)
    s[h] = act ? attnf[(((size_t)b * H_ + h) * N_ + i) * N_ + j] : 0.0f;

  float m1[8];
#pragma unroll
  for (int g = 0; g < 8; ++g) {
    float a = sbl[g];
#pragma unroll
    for (int h = 0; h < 8; ++h) a += s[h] * swl[h * 8 + g];
    m1[g] = a;
  }
  // row max over j (per g)
#pragma unroll
  for (int g = 0; g < 8; ++g) red[g][tid] = act ? m1[g] : -3.0e38f;
  __syncthreads();
  for (int st = 128; st > 0; st >>= 1) {
    if (tid < st)
#pragma unroll
      for (int g = 0; g < 8; ++g) red[g][tid] = fmaxf(red[g][tid], red[g][tid + st]);
    __syncthreads();
  }
  float p[8];
#pragma unroll
  for (int g = 0; g < 8; ++g) p[g] = act ? __expf(m1[g] - red[g][0]) : 0.0f;
  __syncthreads();
  // row sum over j (per g)
#pragma unroll
  for (int g = 0; g < 8; ++g) red[g][tid] = p[g];
  __syncthreads();
  for (int st = 128; st > 0; st >>= 1) {
    if (tid < st)
#pragma unroll
      for (int g = 0; g < 8; ++g) red[g][tid] += red[g][tid + st];
    __syncthreads();
  }
#pragma unroll
  for (int g = 0; g < 8; ++g) p[g] *= 1.0f / red[g][0];

  float a2[8];
#pragma unroll
  for (int g = 0; g < 8; ++g) {
    float a = sbw[g];
#pragma unroll
    for (int h = 0; h < 8; ++h) a += p[h] * sww[h * 8 + g];
    a2[g] = a;
  }
  if (act) {
#pragma unroll
    for (int g = 0; g < 8; ++g) {
      attnf[(((size_t)b * H_ + g) * N_   + i) * N_   + j] = a2[g];         // attn out (fp32)
      attnb[(((size_t)b * H_ + g) * NP16 + i) * NP32 + j] = (__bf16)a2[g]; // padded bf16
    }
  }
}

// --------------------- stage 4: O = attn @ V per (b,h) ----------------------
// One wave per 16x64 tile (full D), A-fragment reused across 4 WMMAs.
__global__ __launch_bounds__(256) void k_av(const __bf16* __restrict__ attnb,
                                            const __bf16* __restrict__ vt,
                                            __bf16* __restrict__ obf) {
  const int lane = threadIdx.x & 31;
  const int gw   = blockIdx.x * 8 + (threadIdx.x >> 5);
  const int bh   = gw / 13;            // 13 i-tiles per (b,h)
  const int ti   = gw % 13;
  const __bf16* abase = attnb + (size_t)bh * NP16 * NP32 + (size_t)ti * 16 * NP32;
  const __bf16* bbase = vt    + (size_t)bh * D_ * NP32;
  v8f acc[4] = {};
#pragma unroll 2
  for (int s = 0; s < NP32 / 32; ++s) { // 7 steps; padded region is zero
    v16bf a = load_a16x32(abase + s * 32, NP32, lane);
#pragma unroll
    for (int t = 0; t < 4; ++t) {
      v16bf b = load_b32x16(bbase + (size_t)t * 16 * NP32 + s * 32, NP32, lane);
      acc[t] = wmma_bf16(a, b, acc[t]);
    }
  }
  const int bi = bh / H_, h = bh % H_;
  const int r0 = (lane >> 4) << 3;
#pragma unroll
  for (int t = 0; t < 4; ++t) {
    const int dc = t * 16 + (lane & 15);
#pragma unroll
    for (int r = 0; r < 8; ++r) {
      const int tr = ti * 16 + r0 + r;
      if (tr < N_)
        obf[((size_t)(bi * N_ + tr)) * C_ + h * D_ + dc] = (__bf16)acc[t][r];
    }
  }
}

// ------------------ stage 5: out = O @ w_proj + b_proj ----------------------
__global__ __launch_bounds__(256) void k_proj(const __bf16* __restrict__ obf,
                                              const __bf16* __restrict__ wT,
                                              const float* __restrict__ bp,
                                              float* __restrict__ out) {
  const int lane = threadIdx.x & 31;
  const int gw   = blockIdx.x * 8 + (threadIdx.x >> 5);
  const int tm   = gw / (C_ / 64);     // 8 col-blocks of 64
  const int tn   = gw % (C_ / 64);
  const __bf16* abase = obf + (size_t)tm * 16 * C_;
  const __bf16* bbase = wT  + (size_t)tn * 64 * C_;
  v8f acc[4] = {};
#pragma unroll 2
  for (int s = 0; s < C_ / 32; ++s) {
    if (s + 1 < C_ / 32) {
      __builtin_prefetch(abase + (s + 1) * 32 + (size_t)(lane & 15) * C_, 0, 3);
      __builtin_prefetch(bbase + (s + 1) * 32 + (size_t)(lane & 15) * C_, 0, 3);
    }
    v16bf a = load_a16x32(abase + s * 32, C_, lane);
#pragma unroll
    for (int t = 0; t < 4; ++t) {
      v16bf b = load_b32x16(bbase + (size_t)t * 16 * C_ + s * 32, C_, lane);
      acc[t] = wmma_bf16(a, b, acc[t]);
    }
  }
  const int r0 = (lane >> 4) << 3;
#pragma unroll
  for (int t = 0; t < 4; ++t) {
    const int col = tn * 64 + t * 16 + (lane & 15);
    const float bias = bp[col];
#pragma unroll
    for (int r = 0; r < 8; ++r)
      out[(size_t)(tm * 16 + r0 + r) * C_ + col] = acc[t][r] + bias;
  }
}

// ------------------------------- launcher -----------------------------------

extern "C" void kernel_launch(void* const* d_in, const int* in_sizes, int n_in,
                              void* d_out, int out_size, void* d_ws, size_t ws_size,
                              hipStream_t stream) {
  const float* x      = (const float*)d_in[0];
  const float* w_qkv  = (const float*)d_in[1];
  const float* b_qkv  = (const float*)d_in[2];
  const float* w_l    = (const float*)d_in[3];
  const float* b_l    = (const float*)d_in[4];
  const float* w_w    = (const float*)d_in[5];
  const float* b_w    = (const float*)d_in[6];
  const float* w_proj = (const float*)d_in[7];
  const float* b_proj = (const float*)d_in[8];

  float* out   = (float*)d_out;
  float* attnf = out + (size_t)M_ * C_;   // attn output region, reused for raw scores

  char* ws = (char*)d_ws;
  size_t off = 0;
  auto take = [&](size_t bytes) -> char* {
    char* p = ws + off;
    off = (off + bytes + 255) & ~(size_t)255;
    return p;
  };
  __bf16* xb     = (__bf16*)take((size_t)M_ * C_ * 2);
  __bf16* wqkvT  = (__bf16*)take((size_t)QKVN * C_ * 2);
  __bf16* wprojT = (__bf16*)take((size_t)C_ * C_ * 2);
  char* zstart = ws + off;
  __bf16* q      = (__bf16*)take((size_t)B_ * H_ * NP16 * D_ * 2);
  __bf16* kmat   = (__bf16*)take((size_t)B_ * H_ * NP16 * D_ * 2);
  __bf16* vt     = (__bf16*)take((size_t)B_ * H_ * D_ * NP32 * 2);
  __bf16* attnb  = (__bf16*)take((size_t)B_ * H_ * NP16 * NP32 * 2);
  char* zend = ws + off;
  __bf16* obf    = (__bf16*)take((size_t)M_ * C_ * 2);
  (void)in_sizes; (void)n_in; (void)out_size; (void)ws_size;

  // 0) zero padded scratch (padding must contribute 0 to WMMA K-sums)
  const size_t n16 = (size_t)(zend - zstart) / 16;
  k_zero<<<2048, 256, 0, stream>>>((uint4*)zstart, n16);

  // 1) convert / transpose operands to bf16
  k_cvt_bf16<<<4096, 256, 0, stream>>>(x, xb, M_ * C_);
  k_transpose_cvt<<<2048, 256, 0, stream>>>(w_qkv, wqkvT, C_, QKVN);
  k_transpose_cvt<<<1024, 256, 0, stream>>>(w_proj, wprojT, C_, C_);

  // 2) QKV GEMM + scatter  (788 * 24 waves, 8 waves/block)
  k_qkv<<<(MT_ * (QKVN / 64)) / 8, 256, 0, stream>>>(xb, wqkvT, b_qkv, q, kmat, vt);

  // 3) scores S = q @ k^T  (512 bh * 169 tiles)
  k_scores<<<(B_ * H_ * 169) / 8, 256, 0, stream>>>(q, kmat, attnf);

  // 4) mix1 -> softmax -> mix2, in place on attn region
  k_mix_softmax<<<B_ * N_, 256, 0, stream>>>(attnf, attnb, w_l, b_l, w_w, b_w);

  // 5) O = attn @ V  (512 bh * 13 waves)
  k_av<<<(B_ * H_ * 13) / 8, 256, 0, stream>>>(attnb, vt, obf);

  // 6) out = O @ w_proj + b_proj  (788 * 8 waves)
  k_proj<<<(MT_ * (C_ / 64)) / 8, 256, 0, stream>>>(obf, wprojT, b_proj, out);
}
